// PreGatingContextualAttention_80693845557344
// MI455X (gfx1250) — matrix operand order repaired and checked by reading, not compiled, not run on hardware
//
#include <hip/hip_runtime.h>
#include <math.h>

#define E_DIM 1024
#define T_DIM 4096
#define S_DIM 4096

typedef __bf16 v16bf __attribute__((ext_vector_type(16)));
typedef float  v8f   __attribute__((ext_vector_type(8)));
typedef unsigned int u32x4 __attribute__((ext_vector_type(4)));
typedef int          i32x8 __attribute__((ext_vector_type(8)));
typedef int          i32x4 __attribute__((ext_vector_type(4)));

#if defined(__HIP_DEVICE_COMPILE__) && __has_builtin(__builtin_amdgcn_tensor_load_to_lds) && __has_builtin(__builtin_amdgcn_s_wait_tensorcnt)
#define HAVE_TDM 1
#else
#define HAVE_TDM 0
#endif

// ---------------- WMMA fragment helpers (CDNA5 16x16x32 bf16 layouts) ----------------
// A-matrix 16x32 bf16 (ISA 7.12.2): lane m = L%16, K-half = 8*(L/16);
// VGPR j holds K pairs {2j,2j+1} for j<4 and {16+2(j-4), ...} for j>=4.
__device__ __forceinline__ v16bf frag_a(const __bf16* s, int row0, int ld, int lane) {
  const int m  = lane & 15;
  const int kh = (lane >> 4) * 8;
  const __bf16* p = s + (row0 + m) * ld;
  v16bf f;
#pragma unroll
  for (int j = 0; j < 8; ++j) {
    const int kb = ((j < 4) ? (2 * j) : (16 + 2 * (j - 4))) + kh;
    f[2 * j]     = p[kb];
    f[2 * j + 1] = p[kb + 1];
  }
  return f;
}
// B-matrix 32x16 bf16: lane n = L%16 holds K = 8*(L/16)+{0..7} and 16+8*(L/16)+{0..7} (same as A).
__device__ __forceinline__ v16bf frag_b(const __bf16* s, int row0, int ld, int lane) {
  const int n  = lane & 15;
  const int kb = (lane >> 4) * 8;
  const __bf16* p = s + (row0 + n) * ld + kb;
  v16bf f;
#pragma unroll
  for (int e = 0; e < 16; ++e) f[e] = p[(e < 8) ? e : (e + 8)];
  return f;
}
__device__ __forceinline__ v8f wmma_bf16(v16bf a, v16bf b, v8f c) {
  // (neg_a, A, neg_b, B, c_mod, C, reuse_a, reuse_b)
  return __builtin_amdgcn_wmma_f32_16x16x32_bf16(false, a, false, b, (short)0, c, false, false);
}
__device__ __forceinline__ float elu1(float x) { return x > 0.f ? x : __expf(x) - 1.f; }
// Branch-free tanh: native v_exp_f32 + fast divide; clamp keeps exp finite.
__device__ __forceinline__ float fast_tanh(float x) {
  const float xc = fminf(fmaxf(x, -10.f), 10.f);
  const float e  = __expf(2.f * xc);
  return __fdividef(e - 1.f, e + 1.f);
}

// ---------------- Generic GEMM: C[M,N] = A[M,K] @ B[N,K]^T (+bias, +ELU) ----------------
#define BM 128
#define BN 128
#define BK 32
#define LDAB 40   // padded bf16 row stride in LDS

template <int EPI>
__global__ void __launch_bounds__(256) gemm_kernel(
    const float* __restrict__ A, const float* __restrict__ B,
    const float* __restrict__ bias, float* __restrict__ C,
    int M, int N, int K)
{
  __shared__ __bf16 As[BM * LDAB];
  __shared__ __bf16 Bs[BN * LDAB];
  const int tid  = threadIdx.x;
  const int lane = tid & 31;
  const int wave = tid >> 5;     // 0..7
  const int wm   = wave >> 1;    // 0..3 : wave-row (32 rows each)
  const int wn   = wave & 1;     // 0..1 : wave-col (64 cols each)
  const int bM   = blockIdx.y * BM;
  const int bN   = blockIdx.x * BN;

  v8f acc[2][4] = {};

  for (int kt = 0; kt < K; kt += BK) {
    // Stage 128x32 fp32 -> bf16 tiles for A and B (4 float4 per thread each)
#pragma unroll
    for (int i = 0; i < 4; ++i) {
      const int idx = tid + i * 256;       // 0..1023
      const int r   = idx >> 3;            // 0..127
      const int c4  = (idx & 7) * 4;       // 0,4,..,28
      const float4 va = *(const float4*)(A + (size_t)(bM + r) * K + kt + c4);
      __bf16* da = As + r * LDAB + c4;
      da[0] = (__bf16)va.x; da[1] = (__bf16)va.y; da[2] = (__bf16)va.z; da[3] = (__bf16)va.w;
      const float4 vb = *(const float4*)(B + (size_t)(bN + r) * K + kt + c4);
      __bf16* db = Bs + r * LDAB + c4;
      db[0] = (__bf16)vb.x; db[1] = (__bf16)vb.y; db[2] = (__bf16)vb.z; db[3] = (__bf16)vb.w;
      if (kt + BK < K) {  // prefetch next K tile -> global_prefetch_b8
        __builtin_prefetch(A + (size_t)(bM + r) * K + kt + BK + c4, 0, 0);
        __builtin_prefetch(B + (size_t)(bN + r) * K + kt + BK + c4, 0, 0);
      }
    }
    __syncthreads();

    v16bf af[2], bfr[4];
#pragma unroll
    for (int ti = 0; ti < 2; ++ti) af[ti] = frag_a(As, wm * 32 + ti * 16, LDAB, lane);
#pragma unroll
    for (int tj = 0; tj < 4; ++tj) bfr[tj] = frag_b(Bs, wn * 64 + tj * 16, LDAB, lane);
#pragma unroll
    for (int ti = 0; ti < 2; ++ti)
#pragma unroll
      for (int tj = 0; tj < 4; ++tj)
        acc[ti][tj] = wmma_bf16(af[ti], bfr[tj], acc[ti][tj]);
    __syncthreads();
  }

  // Epilogue: C/D layout — VGPR r, lanes 0-15: M=r, lanes 16-31: M=r+8; N=lane%16
  const int mrow = (lane >> 4) * 8;
  const int ncol = lane & 15;
#pragma unroll
  for (int ti = 0; ti < 2; ++ti)
#pragma unroll
    for (int tj = 0; tj < 4; ++tj) {
      const int col = bN + wn * 64 + tj * 16 + ncol;
      const float bb = bias ? bias[col] : 0.f;
#pragma unroll
      for (int r = 0; r < 8; ++r) {
        const int row = bM + wm * 32 + ti * 16 + mrow + r;
        float v = acc[ti][tj][r] + bb;
        if (EPI == 1) v = elu1(v);
        C[(size_t)row * N + col] = v;
      }
    }
}

// ------- Fused dual GEMM: logits = (q@k.T/32) * ((tanh(q)@tanh(k).T + 1)*0.5) -------
#define FBM 64
#define FBN 128
__global__ void __launch_bounds__(256) scores_kernel(
    const float* __restrict__ q, const float* __restrict__ k,
    float* __restrict__ logits, int Tq, int Sk, int K)
{
  __shared__ __bf16 Aq[FBM * LDAB];
  __shared__ __bf16 At[FBM * LDAB];
  __shared__ __bf16 Bk[FBN * LDAB];
  __shared__ __bf16 Bt[FBN * LDAB];
  const int tid  = threadIdx.x;
  const int lane = tid & 31;
  const int wave = tid >> 5;
  const int wm   = wave >> 2;   // 0..1 (32 rows each)
  const int wn   = wave & 3;    // 0..3 (32 cols each)
  const int bM   = blockIdx.y * FBM;
  const int bN   = blockIdx.x * FBN;

  v8f acc1[2][2] = {};
  v8f acc2[2][2] = {};

  for (int kt = 0; kt < K; kt += BK) {
#pragma unroll
    for (int i = 0; i < 2; ++i) {          // A: 64 rows x 8 float4
      const int idx = tid + i * 256;
      const int r   = idx >> 3;
      const int c4  = (idx & 7) * 4;
      const float4 va = *(const float4*)(q + (size_t)(bM + r) * K + kt + c4);
      __bf16* dq = Aq + r * LDAB + c4;
      __bf16* dt = At + r * LDAB + c4;
      dq[0] = (__bf16)va.x; dq[1] = (__bf16)va.y; dq[2] = (__bf16)va.z; dq[3] = (__bf16)va.w;
      dt[0] = (__bf16)fast_tanh(va.x); dt[1] = (__bf16)fast_tanh(va.y);
      dt[2] = (__bf16)fast_tanh(va.z); dt[3] = (__bf16)fast_tanh(va.w);
    }
#pragma unroll
    for (int i = 0; i < 4; ++i) {          // B: 128 rows x 8 float4
      const int idx = tid + i * 256;
      const int r   = idx >> 3;
      const int c4  = (idx & 7) * 4;
      const float4 vb = *(const float4*)(k + (size_t)(bN + r) * K + kt + c4);
      __bf16* dk = Bk + r * LDAB + c4;
      __bf16* dt = Bt + r * LDAB + c4;
      dk[0] = (__bf16)vb.x; dk[1] = (__bf16)vb.y; dk[2] = (__bf16)vb.z; dk[3] = (__bf16)vb.w;
      dt[0] = (__bf16)fast_tanh(vb.x); dt[1] = (__bf16)fast_tanh(vb.y);
      dt[2] = (__bf16)fast_tanh(vb.z); dt[3] = (__bf16)fast_tanh(vb.w);
    }
    __syncthreads();

    v16bf aq[2], at[2], bk_[2], bt[2];
#pragma unroll
    for (int ti = 0; ti < 2; ++ti) {
      aq[ti] = frag_a(Aq, wm * 32 + ti * 16, LDAB, lane);
      at[ti] = frag_a(At, wm * 32 + ti * 16, LDAB, lane);
    }
#pragma unroll
    for (int tj = 0; tj < 2; ++tj) {
      bk_[tj] = frag_b(Bk, wn * 32 + tj * 16, LDAB, lane);
      bt[tj]  = frag_b(Bt, wn * 32 + tj * 16, LDAB, lane);
    }
#pragma unroll
    for (int ti = 0; ti < 2; ++ti)
#pragma unroll
      for (int tj = 0; tj < 2; ++tj) {
        acc1[ti][tj] = wmma_bf16(aq[ti], bk_[tj], acc1[ti][tj]);
        acc2[ti][tj] = wmma_bf16(at[ti], bt[tj],  acc2[ti][tj]);
      }
    __syncthreads();
  }

  const int mrow = (lane >> 4) * 8;
  const int ncol = lane & 15;
#pragma unroll
  for (int ti = 0; ti < 2; ++ti)
#pragma unroll
    for (int tj = 0; tj < 2; ++tj)
#pragma unroll
      for (int r = 0; r < 8; ++r) {
        const int row = bM + wm * 32 + ti * 16 + mrow + r;
        const int col = bN + wn * 32 + tj * 16 + ncol;
        const float s1 = acc1[ti][tj][r];
        const float s2 = acc2[ti][tj][r];
        logits[(size_t)row * Sk + col] = (s1 * 0.03125f) * ((s2 + 1.f) * 0.5f);
      }
}

// ---------------- In-place row softmax (row length 4096, 256 threads) ----------------
// The 16KB row is pulled into LDS with the Tensor Data Mover (1-D tile descriptor),
// then reduced from LDS; result written back to global.
__global__ void __launch_bounds__(256) softmax_kernel(float* __restrict__ x, int Sk) {
  __shared__ float rowbuf[S_DIM];   // 16 KB
  __shared__ float red[256];
  const int row = blockIdx.x;
  const int tid = threadIdx.x;
  float* p = x + (size_t)row * Sk;

#if HAVE_TDM
  if (tid < 32) {   // TDM is a wave-level op (EXEC ignored): issue from wave 0 only
    const unsigned           lds = (unsigned)(uintptr_t)rowbuf;       // low 32b = LDS byte offset
    const unsigned long long ga  = (unsigned long long)(uintptr_t)p;  // 57-bit global address
    // D# group 0: count=1 | lds_addr | global_addr | type=2 ("image")
    u32x4 g0;
    g0.x = 1u;
    g0.y = lds;
    g0.z = (unsigned)(ga & 0xffffffffu);
    g0.w = (unsigned)((ga >> 32) & 0x01ffffffu) | (2u << 30);
    // D# group 1: data_size=4B (2), tensor_dim0=tile_dim0=Sk, dim1=1, stride0=Sk
    i32x8 g1;
    g1[0] = (2 << 16);                                   // data_size=2 (4 bytes)
    g1[1] = (int)((unsigned)(Sk & 0xffff) << 16);        // tensor_dim0[15:0] in bits 63:48
    g1[2] = (int)(((unsigned)Sk >> 16) | (1u << 16));    // tensor_dim0[31:16] | tensor_dim1=1
    g1[3] = (int)((unsigned)(Sk & 0xffff) << 16);        // tile_dim0 in bits 127:112
    g1[4] = 1;                                           // tile_dim1=1
    g1[5] = Sk;                                          // tensor_dim0_stride[31:0]
    g1[6] = 0;
    g1[7] = 0;
    i32x4 g2 = {0, 0, 0, 0};
    i32x4 g3 = {0, 0, 0, 0};
    i32x8 g4 = {0, 0, 0, 0, 0, 0, 0, 0};                 // extra group (6-arg toolchain form)
    __builtin_amdgcn_tensor_load_to_lds(g0, g1, g2, g3, g4, 0);
    __builtin_amdgcn_s_wait_tensorcnt(0);
  }
  __syncthreads();
#else
#pragma unroll
  for (int i = 0; i < 16; ++i) rowbuf[tid + i * 256] = p[tid + i * 256];
  __syncthreads();
#endif

  float v[16];
  float mx = -3.4e38f;
#pragma unroll
  for (int i = 0; i < 16; ++i) { v[i] = rowbuf[tid + i * 256]; mx = fmaxf(mx, v[i]); }
  red[tid] = mx; __syncthreads();
  for (int s = 128; s > 0; s >>= 1) { if (tid < s) red[tid] = fmaxf(red[tid], red[tid + s]); __syncthreads(); }
  mx = red[0]; __syncthreads();
  float sum = 0.f;
#pragma unroll
  for (int i = 0; i < 16; ++i) { v[i] = __expf(v[i] - mx); sum += v[i]; }
  red[tid] = sum; __syncthreads();
  for (int s = 128; s > 0; s >>= 1) { if (tid < s) red[tid] += red[tid + s]; __syncthreads(); }
  const float inv = 1.f / red[0];
#pragma unroll
  for (int i = 0; i < 16; ++i) p[tid + i * 256] = v[i] * inv;
}

// ---------------- LayerNorm over rows of length 1024 ----------------
__global__ void __launch_bounds__(256) ln_kernel(
    const float* __restrict__ x, const float* __restrict__ w,
    const float* __restrict__ b, float* __restrict__ y, int N)
{
  __shared__ float red[256];
  const int row = blockIdx.x;
  const int tid = threadIdx.x;
  const float* p = x + (size_t)row * N;
  float v[4];
  float s = 0.f;
#pragma unroll
  for (int i = 0; i < 4; ++i) { v[i] = p[tid + i * 256]; s += v[i]; }
  red[tid] = s; __syncthreads();
  for (int t = 128; t > 0; t >>= 1) { if (tid < t) red[tid] += red[tid + t]; __syncthreads(); }
  const float mu = red[0] * (1.f / 1024.f);
  __syncthreads();
  float d = 0.f;
#pragma unroll
  for (int i = 0; i < 4; ++i) { const float dd = v[i] - mu; d += dd * dd; }
  red[tid] = d; __syncthreads();
  for (int t = 128; t > 0; t >>= 1) { if (tid < t) red[tid] += red[tid + t]; __syncthreads(); }
  const float rstd = rsqrtf(red[0] * (1.f / 1024.f) + 1e-5f);
#pragma unroll
  for (int i = 0; i < 4; ++i) {
    const int c = tid + i * 256;
    y[(size_t)row * N + c] = (v[i] - mu) * rstd * w[c] + b[c];
  }
}

// ---------------- fp32 transpose (R x C -> C x R), 32x32 tiles ----------------
__global__ void __launch_bounds__(256) transpose_kernel(
    const float* __restrict__ src, float* __restrict__ dst, int R, int Cc)
{
  __shared__ float tile[32][33];
  const int bx = blockIdx.x * 32;  // col base in src
  const int by = blockIdx.y * 32;  // row base in src
  const int tx = threadIdx.x & 31;
  const int ty = threadIdx.x >> 5; // 0..7
#pragma unroll
  for (int i = 0; i < 4; ++i) {
    const int r = ty + i * 8;
    tile[r][tx] = src[(size_t)(by + r) * Cc + bx + tx];
  }
  __syncthreads();
#pragma unroll
  for (int i = 0; i < 4; ++i) {
    const int r = ty + i * 8;
    dst[(size_t)(bx + r) * R + by + tx] = tile[tx][r];
  }
}

// ---------------- elementwise kernels (float4 vectorized) ----------------
__global__ void __launch_bounds__(256) ew_add_elu_kernel(float4* a, const float4* b) {
  const int i = blockIdx.x * 256 + threadIdx.x;
  float4 x = a[i]; const float4 y = b[i];
  x.x = elu1(x.x + y.x); x.y = elu1(x.y + y.y);
  x.z = elu1(x.z + y.z); x.w = elu1(x.w + y.w);
  a[i] = x;
}
__global__ void __launch_bounds__(256) ew_elu_kernel(float4* a) {
  const int i = blockIdx.x * 256 + threadIdx.x;
  float4 x = a[i];
  x.x = elu1(x.x); x.y = elu1(x.y); x.z = elu1(x.z); x.w = elu1(x.w);
  a[i] = x;
}
__global__ void __launch_bounds__(256) ew_mul_kernel(const float4* a, const float4* b, float4* o) {
  const int i = blockIdx.x * 256 + threadIdx.x;
  const float4 x = a[i], y = b[i];
  o[i] = make_float4(x.x * y.x, x.y * y.y, x.z * y.z, x.w * y.w);
}
__global__ void __launch_bounds__(256) ew_add_kernel(const float4* a, const float4* b, float4* o) {
  const int i = blockIdx.x * 256 + threadIdx.x;
  const float4 x = a[i], y = b[i];
  o[i] = make_float4(x.x + y.x, x.y + y.y, x.z + y.z, x.w + y.w);
}

// ---------------- driver ----------------
extern "C" void kernel_launch(void* const* d_in, const int* in_sizes, int n_in,
                              void* d_out, int out_size, void* d_ws, size_t ws_size,
                              hipStream_t stream) {
  (void)in_sizes; (void)n_in; (void)out_size; (void)ws_size;
  const int E = E_DIM, T = T_DIM, S = S_DIM;
  const float* query = (const float*)d_in[0];
  const float* key_  = (const float*)d_in[1];
  const float* value = (const float*)d_in[2];
  const float* ipw   = (const float*)d_in[3];
  const float* ipb   = (const float*)d_in[4];
  const float* out_w = (const float*)d_in[5];
  const float* out_b = (const float*)d_in[6];
  const float* fc1_w = (const float*)d_in[7];
  const float* fc1_b = (const float*)d_in[8];
  const float* fc2_w = (const float*)d_in[9];
  const float* fc2_b = (const float*)d_in[10];
  const float* fc3_w = (const float*)d_in[11];
  const float* fc3_b = (const float*)d_in[12];
  const float* lnG_w = (const float*)d_in[13];
  const float* lnG_b = (const float*)d_in[14];
  const float* lnE_w = (const float*)d_in[15];
  const float* lnE_b = (const float*)d_in[16];
  const float* fcc_w = (const float*)d_in[17];
  const float* fcc_b = (const float*)d_in[18];

  const float* wq = ipw;
  const float* wk = ipw + (size_t)E * E;
  const float* wv = ipw + (size_t)2 * E * E;
  const float* bq = ipb;
  const float* bk = ipb + E;
  const float* bv = ipb + 2 * E;

  float* out0 = (float*)d_out;          // [T,E]  attn_out + C
  float* attn = out0 + (size_t)T * E;   // [T,S]  logits -> softmax in place

  char* wsb = (char*)d_ws;
  const size_t MB = (size_t)1 << 20;
  float* qb    = (float*)(wsb +  0 * MB);  // q fp32   (16MB), live whole pass
  float* kb    = (float*)(wsb + 16 * MB);  // k fp32   (16MB), reused later
  float* vb    = (float*)(wsb + 32 * MB);  // v fp32   (16MB), reused later
  float* vT    = (float*)(wsb + 48 * MB);  // v^T fp32 (16MB), reused later
  float* attnv = (float*)(wsb + 64 * MB);  // attn@v   (16MB), reused later
  float* attno = (float*)(wsb + 80 * MB);  // attn_out (16MB)
  // reuse schedule (stream-ordered):
  float* g1  = kb;      // after scores_kernel consumed k
  float* g2  = vb;      // after transpose consumed v
  float* G   = vT;      // after attn@v consumed vT
  float* e1  = kb;      // after LN consumed g1
  float* Eg  = vb;      // after add_elu consumed g2
  float* gEg = kb;      // after LN consumed e1
  float* Cb  = attnv;   // after out-proj consumed attnv

  const dim3 blk(256);
  const dim3 gproj(E / BN, T / BM);     // (8,32)
  const int n4 = T * E / 4;             // float4 count
  const dim3 gew(n4 / 256);

  // --- attention path ---
  gemm_kernel<0><<<gproj, blk, 0, stream>>>(query, wq, bq, qb, T, E, E);
  gemm_kernel<0><<<gproj, blk, 0, stream>>>(key_,  wk, bk, kb, T, E, E);
  gemm_kernel<0><<<gproj, blk, 0, stream>>>(value, wv, bv, vb, T, E, E);
  transpose_kernel<<<dim3(E / 32, S / 32), blk, 0, stream>>>(vb, vT, S, E);
  scores_kernel<<<dim3(S / FBN, T / FBM), blk, 0, stream>>>(qb, kb, attn, T, S, E);
  softmax_kernel<<<dim3(T), blk, 0, stream>>>(attn, S);
  gemm_kernel<0><<<gproj, blk, 0, stream>>>(attn, vT, nullptr, attnv, T, E, S);
  gemm_kernel<0><<<gproj, blk, 0, stream>>>(attnv, out_w, out_b, attno, T, E, E);

  // --- gating path ---
  gemm_kernel<1><<<gproj, blk, 0, stream>>>(query, fc1_w, fc1_b, g1, T, E, E);
  gemm_kernel<1><<<gproj, blk, 0, stream>>>(qb,    fc2_w, fc2_b, g2, T, E, E);
  ew_add_elu_kernel<<<gew, blk, 0, stream>>>((float4*)g1, (const float4*)g2);
  ln_kernel<<<dim3(T), blk, 0, stream>>>(g1, lnG_w, lnG_b, G, E);
  gemm_kernel<1><<<gproj, blk, 0, stream>>>(qb, fc3_w, fc3_b, e1, T, E, E);
  ew_elu_kernel<<<gew, blk, 0, stream>>>((float4*)e1);
  ln_kernel<<<dim3(T), blk, 0, stream>>>(e1, lnE_w, lnE_b, Eg, E);
  ew_mul_kernel<<<gew, blk, 0, stream>>>((const float4*)G, (const float4*)Eg, (float4*)gEg);
  gemm_kernel<1><<<gproj, blk, 0, stream>>>(gEg, fcc_w, fcc_b, Cb, T, E, E);

  // --- combine ---
  ew_add_kernel<<<gew, blk, 0, stream>>>((const float4*)attno, (const float4*)Cb, (float4*)out0);
}
